// HopfieldLayer_8134668059251
// MI455X (gfx1250) — matrix operand order repaired and checked by reading, not compiled, not run on hardware
//
#include <hip/hip_runtime.h>
#include <math.h>

// ---------------------------------------------------------------------------
// Fused Hopfield layer for gfx1250 (MI455X), wave32 + v_wmma_f32_16x16x32_bf16
//   Pre-pass: Wl, Wc converted f32 -> bf16 ONCE into d_ws (12.6 MB)
//   scores  = x @ Wl^T * beta        [16 tok x 4096]  (kept in LDS, bf16)
//   P       = softmax(scores)        (unnormalized exp in LDS, 1/sum folded out)
//   content = P @ Wc^T               [16 tok x 768]
// One block = 16 tokens, 256 threads = 8 wave32s.
// Hot loop: ds_load_b128 (A, LDS) + clause'd global_load_b128 (B, bf16 via L2)
//           + v_wmma_f32_16x16x32_bf16, overlapped via split wait counters.
// ---------------------------------------------------------------------------

typedef __bf16 v16bf __attribute__((ext_vector_type(16)));
typedef float  v8f   __attribute__((ext_vector_type(8)));

#define DIM     768
#define NPROTO  4096
#define XS      784     // x-tile LDS row stride (bf16 elems), +16 pad
#define SS      4112    // score LDS row stride (bf16 elems), +16 pad
#define BETA    0.03608439182435161f   // 1/sqrt(768)

// --- conversions -----------------------------------------------------------

// Exact RNE pack (one-time weight pre-pass + x-tile staging; fully amortized)
__device__ __forceinline__ unsigned int pk_bf16_rne(float a, float b) {
    union { float f; unsigned int u; } x, y;
    x.f = a; y.f = b;
    unsigned int lo = (x.u + 0x7FFFu + ((x.u >> 16) & 1u)) >> 16;
    unsigned int hi = (y.u + 0x7FFFu + ((y.u >> 16) & 1u)) >> 16;
    return lo | (hi << 16);
}

// Truncating pack (fallback hot loop only; folds toward v_perm_b32)
__device__ __forceinline__ unsigned int pk_bf16_trunc(float a, float b) {
    union { float f; unsigned int u; } x, y;
    x.f = a; y.f = b;
    return (y.u & 0xFFFF0000u) | (x.u >> 16);
}

// Round-half-up scalar conversion (score/exp stores: 2 VALU instead of ~5;
// within 1/2 ulp of RNE, feeds a softmax so the tiny bias is irrelevant)
__device__ __forceinline__ unsigned short f32_to_bf16_fast(float a) {
    union { float f; unsigned int u; } x; x.f = a;
    return (unsigned short)((x.u + 0x8000u) >> 16);
}

__device__ __forceinline__ float bf16_to_f32(unsigned short h) {
    union { unsigned int u; float f; } x; x.u = ((unsigned int)h) << 16;
    return x.f;
}

// --- WMMA operand loaders --------------------------------------------------

// A-operand (16x32 bf16) from an LDS row-major tile.
// m = lane&15; lanes 0-15 hold K [kb..kb+15],
// lanes 16-31 hold K [kb+16..kb+31] (one contiguous 32B run).
__device__ __forceinline__ v16bf loadA_lds(const unsigned short* row,
                                           int kb, int kloa) {
    union { uint4 q[2]; v16bf v; } au;
    au.q[0] = *(const uint4*)(row + kb + kloa);
    au.q[1] = *(const uint4*)(row + kb + kloa + 8);
    return au.v;
}

// B-operand (32x16 bf16) from a pre-converted bf16 global row.
// n = lane&15; lanes 0-15 hold K [kb..kb+15], lanes 16-31 [kb+16..kb+31]:
// one contiguous run of 16 bf16 = 32 B = two b128 loads, no VALU.
__device__ __forceinline__ v16bf loadB_bf16(const unsigned short* row,
                                            int kb, int klob) {
    union { uint4 q[2]; v16bf v; } bu;
    bu.q[0] = *(const uint4*)(row + kb + klob);
    bu.q[1] = *(const uint4*)(row + kb + klob + 8);
    return bu.v;
}

// B-operand from f32 global row (fallback path, truncating pack).
__device__ __forceinline__ v16bf loadB_f32(const float* row, int kb, int klob) {
    const float4* p = (const float4*)(row + kb + klob);
    float4 f0 = p[0], f1 = p[1], f2 = p[2], f3 = p[3];
    union { unsigned int u[8]; v16bf v; } bu;
    bu.u[0] = pk_bf16_trunc(f0.x, f0.y); bu.u[1] = pk_bf16_trunc(f0.z, f0.w);
    bu.u[2] = pk_bf16_trunc(f1.x, f1.y); bu.u[3] = pk_bf16_trunc(f1.z, f1.w);
    bu.u[4] = pk_bf16_trunc(f2.x, f2.y); bu.u[5] = pk_bf16_trunc(f2.z, f2.w);
    bu.u[6] = pk_bf16_trunc(f3.x, f3.y); bu.u[7] = pk_bf16_trunc(f3.z, f3.w);
    return bu.v;
}

// --- one-time weight conversion kernel -------------------------------------

__global__ __launch_bounds__(256)
void cvt_f32_to_bf16_kernel(const float* __restrict__ src,
                            unsigned short* __restrict__ dst, int nquads) {
    int i = blockIdx.x * blockDim.x + threadIdx.x;
    int stride = gridDim.x * blockDim.x;
    for (; i < nquads; i += stride) {
        float4 v = ((const float4*)src)[i];
        uint2 p;
        p.x = pk_bf16_rne(v.x, v.y);
        p.y = pk_bf16_rne(v.z, v.w);
        ((uint2*)dst)[i] = p;
    }
}

// --- fused kernel ----------------------------------------------------------

template <bool BF16W>
__global__ __launch_bounds__(256)
void hopfield_fused_kernel(const float* __restrict__ xg,
                           const float* __restrict__ wl,            // [4096,768] f32
                           const float* __restrict__ wc,            // [768,4096] f32
                           const unsigned short* __restrict__ wlb,  // [4096,768] bf16
                           const unsigned short* __restrict__ wcb,  // [768,4096] bf16
                           float* __restrict__ out) {               // [16384,768]
    __shared__ __align__(16) unsigned short xs[16 * XS];   // x tile, bf16
    __shared__ __align__(16) unsigned short sc[16 * SS];   // scores/P, bf16
    __shared__ float red[16][16];
    __shared__ float rowmax[16];
    __shared__ float rowinv[16];

    const int tid  = threadIdx.x;
    const int lane = tid & 31;
    const int wave = tid >> 5;
    const int n16  = lane & 15;           // N index within a 16-wide tile
    const int hi   = lane >> 4;           // lane-half select
    const int kloa = hi * 16;             // A-operand K sub-offset
    const int klob = hi * 16;             // B-operand K sub-offset
    const long tok0 = (long)blockIdx.x * 16;

    // ---------------- Phase 0: x tile -> LDS (f32 -> bf16) ----------------
    for (int i = tid; i < (16 * DIM) / 4; i += 256) {
        int fidx = i * 4;
        int row  = fidx / DIM;
        int col  = fidx % DIM;
        float4 v = *(const float4*)(xg + (tok0 + row) * DIM + col);
        uint2 p;
        p.x = pk_bf16_rne(v.x, v.y);
        p.y = pk_bf16_rne(v.z, v.w);
        *(uint2*)(&xs[row * XS + col]) = p;
    }
    __syncthreads();

    // ---------------- Phase 1: scores = x @ Wl^T * beta -> LDS ------------
    // Each wave: 32 prototype tiles (512 protos), in 8 groups of 4 so the
    // A-operand LDS read is amortized over 4 WMMAs per K-step.
    const unsigned short* xrow = &xs[(lane & 15) * XS];
#pragma unroll 1
    for (int g = 0; g < 8; ++g) {
        const int tbase = wave * 512 + g * 64;     // first proto of group
        v8f acc[4] = {v8f{}, v8f{}, v8f{}, v8f{}};
#pragma unroll 1
        for (int kb = 0; kb < DIM; kb += 32) {
            v16bf a = loadA_lds(xrow, kb, kloa);
#pragma unroll
            for (int tt = 0; tt < 4; ++tt) {
                const long prow = (long)(tbase + tt * 16 + n16) * DIM;
                v16bf b = BF16W ? loadB_bf16(wlb + prow, kb, klob)
                                : loadB_f32(wl + prow, kb, klob);
                acc[tt] = __builtin_amdgcn_wmma_f32_16x16x32_bf16(
                    false, a, false, b, (short)0, acc[tt], false, false);
            }
        }
        // C layout: VGPR v -> row m = v + 8*hi, col = n16. Scale by beta.
#pragma unroll
        for (int tt = 0; tt < 4; ++tt) {
            const int pcol = tbase + tt * 16 + n16;
#pragma unroll
            for (int v = 0; v < 8; ++v) {
                int m = v + 8 * hi;
                sc[m * SS + pcol] = f32_to_bf16_fast(acc[tt][v] * BETA);
            }
        }
    }
    __syncthreads();

    // ---------------- Phase 2: softmax over 4096 (in LDS) -----------------
    // 16 threads per row; store UNNORMALIZED exp(s - max); fold 1/sum into
    // the final output store (softmax is row-linear).
    {
        const int r = tid >> 4;
        const int j = tid & 15;
        unsigned short* srow = &sc[r * SS];

        float mmax = -3.0e38f;
        for (int c = j; c < NPROTO; c += 16)
            mmax = fmaxf(mmax, bf16_to_f32(srow[c]));
        red[r][j] = mmax;
        __syncthreads();
        if (j == 0) {
            float m = red[r][0];
#pragma unroll
            for (int jj = 1; jj < 16; ++jj) m = fmaxf(m, red[r][jj]);
            rowmax[r] = m;
        }
        __syncthreads();

        const float rm = rowmax[r];
        float psum = 0.0f;
        for (int c = j; c < NPROTO; c += 16) {
            float e = __expf(bf16_to_f32(srow[c]) - rm);
            srow[c] = f32_to_bf16_fast(e);
            psum += e;
        }
        red[r][j] = psum;
        __syncthreads();
        if (j == 0) {
            float s = 0.0f;
#pragma unroll
            for (int jj = 0; jj < 16; ++jj) s += red[r][jj];
            rowinv[r] = 1.0f / s;
        }
        __syncthreads();
    }

    // ---------------- Phase 3: content = P @ Wc^T -------------------------
    // Each wave owns 96 output dims (6 N-tiles); K = 4096 in steps of 32.
    {
        const int colbase = wave * 96;
        const unsigned short* prow = &sc[(lane & 15) * SS];
        v8f acc[6] = {v8f{}, v8f{}, v8f{}, v8f{}, v8f{}, v8f{}};
#pragma unroll 1
        for (int kb = 0; kb < NPROTO; kb += 32) {
            v16bf a = loadA_lds(prow, kb, kloa);
#pragma unroll
            for (int t = 0; t < 6; ++t) {
                const long nrow = (long)(colbase + t * 16 + n16) * NPROTO;
                v16bf b = BF16W ? loadB_bf16(wcb + nrow, kb, klob)
                                : loadB_f32(wc + nrow, kb, klob);
                acc[t] = __builtin_amdgcn_wmma_f32_16x16x32_bf16(
                    false, a, false, b, (short)0, acc[t], false, false);
            }
        }
        float ril[8];
#pragma unroll
        for (int v = 0; v < 8; ++v) ril[v] = rowinv[v + 8 * hi];
#pragma unroll
        for (int t = 0; t < 6; ++t) {
            const int ncol = colbase + t * 16 + n16;
#pragma unroll
            for (int v = 0; v < 8; ++v) {
                long tokrow = tok0 + v + 8 * hi;
                out[tokrow * DIM + ncol] = acc[t][v] * ril[v];
            }
        }
    }
}

// --- host launcher ---------------------------------------------------------

extern "C" void kernel_launch(void* const* d_in, const int* in_sizes, int n_in,
                              void* d_out, int out_size, void* d_ws, size_t ws_size,
                              hipStream_t stream) {
    (void)in_sizes; (void)n_in; (void)out_size;
    const float* x  = (const float*)d_in[0];   // [32,512,768]
    const float* wl = (const float*)d_in[1];   // [4096,768]
    const float* wc = (const float*)d_in[2];   // [768,4096]
    float* out = (float*)d_out;                // [32,512,768]

    const size_t n_wl = (size_t)NPROTO * DIM;       // 3,145,728 elems
    const size_t n_wc = (size_t)DIM * NPROTO;       // 3,145,728 elems
    const size_t need = (n_wl + n_wc) * sizeof(unsigned short);  // ~12.6 MB

    if (ws_size >= need) {
        // Fast path: one-time f32->bf16 weight conversion into scratch, then
        // the fused kernel streams bf16 weights straight into WMMA B operands.
        unsigned short* wlb = (unsigned short*)d_ws;
        unsigned short* wcb = wlb + n_wl;
        cvt_f32_to_bf16_kernel<<<dim3(1024), dim3(256), 0, stream>>>(
            wl, wlb, (int)(n_wl / 4));
        cvt_f32_to_bf16_kernel<<<dim3(1024), dim3(256), 0, stream>>>(
            wc, wcb, (int)(n_wc / 4));
        hopfield_fused_kernel<true><<<dim3(1024), dim3(256), 0, stream>>>(
            x, wl, wc, wlb, wcb, out);
    } else {
        // Fallback: convert weights on the fly inside the hot loop.
        hopfield_fused_kernel<false><<<dim3(1024), dim3(256), 0, stream>>>(
            x, wl, wc, nullptr, nullptr, out);
    }
}
